// FusionMamba_8641474199634
// MI455X (gfx1250) — matrix operand order repaired and checked by reading, not compiled, not run on hardware
//
#include <hip/hip_runtime.h>
#include <hip/hip_bf16.h>

typedef _Float16 v16h __attribute__((ext_vector_type(16)));
typedef _Float16 v8h  __attribute__((ext_vector_type(8)));
typedef float    v8f  __attribute__((ext_vector_type(8)));

// Problem constants (from reference)
constexpr int Bb    = 4;
constexpr int DIMC  = 128;
constexpr int HW    = 48 * 48;        // 2304
constexpr int Lseq  = 2 * HW;         // 4608
constexpr int DIN   = 256;            // d_inner
constexpr int DST   = 16;             // d_state
constexpr int DTR   = 8;              // dt_rank
constexpr int NPROJ = DTR + 2 * DST;  // 40
constexpr int NPAD  = 48;             // x_proj weights padded to 48 rows
constexpr int M_TOK = Bb * Lseq;      // 18432
constexpr int NCHUNK = 48;
constexpr int CLEN   = Lseq / NCHUNK; // 96

__device__ __forceinline__ float siluf(float x) {
  return x / (1.f + __expf(-x));
}

// ---------------------------------------------------------------------------
// Weight f32 -> f16 conversion (tiny). x_proj weights are padded to 48 rows
// with zeros so the GEMM k-loop needs no load guard (zero rows contribute 0).
// ---------------------------------------------------------------------------
__global__ void cvt_weights_k(const float* __restrict__ win,  // [512,128]
                              const float* __restrict__ wxp,  // [40,256]
                              const float* __restrict__ wout, // [128,256]
                              _Float16* __restrict__ a,
                              _Float16* __restrict__ b,       // [48,256] padded
                              _Float16* __restrict__ c) {
  int i = blockIdx.x * blockDim.x + threadIdx.x;
  constexpr int SZ_IN  = 512 * 128;
  constexpr int SZ_XP  = NPAD * 256;    // padded
  constexpr int SZ_OUT = 128 * 256;
  if (i < SZ_IN) {
    a[i] = (_Float16)win[i];
  } else if (i < SZ_IN + SZ_XP) {
    int j = i - SZ_IN;
    b[j] = (j < NPROJ * 256) ? (_Float16)wxp[j] : (_Float16)0.f;
  } else if (i < SZ_IN + SZ_XP + SZ_OUT) {
    int j = i - (SZ_IN + SZ_XP);
    c[j] = (_Float16)wout[j];
  }
}

// ---------------------------------------------------------------------------
// Build cat [b, L, 128] (f32, residual) and LayerNorm(cat) (f16, WMMA A op).
// One wave32 per token, 4 channels per lane.
// ---------------------------------------------------------------------------
__global__ void ln_cat_k(const float* __restrict__ pan, const float* __restrict__ ms,
                         const float* __restrict__ lw, const float* __restrict__ lb,
                         float* __restrict__ cat, _Float16* __restrict__ lnx) {
  int tok  = blockIdx.x * 8 + (threadIdx.x >> 5);
  int lane = threadIdx.x & 31;
  if (tok >= M_TOK) return;
  int b = tok / Lseq;
  int t = tok - b * Lseq;
  const float* src = (t < HW) ? pan : ms;
  int s = (t < HW) ? t : (t - HW);

  float v[4];
  float sum = 0.f;
#pragma unroll
  for (int j = 0; j < 4; ++j) {
    int c = lane + 32 * j;
    v[j] = src[(size_t)(b * DIMC + c) * HW + s];
    sum += v[j];
  }
#pragma unroll
  for (int o = 16; o >= 1; o >>= 1) sum += __shfl_xor(sum, o, 32);
  float mean = sum * (1.f / DIMC);
  float var = 0.f;
#pragma unroll
  for (int j = 0; j < 4; ++j) {
    float d = v[j] - mean;
    var += d * d;
  }
#pragma unroll
  for (int o = 16; o >= 1; o >>= 1) var += __shfl_xor(var, o, 32);
  float inv = rsqrtf(var * (1.f / DIMC) + 1e-5f);

  size_t base = (size_t)tok * DIMC;
#pragma unroll
  for (int j = 0; j < 4; ++j) {
    int c = lane + 32 * j;
    cat[base + c] = v[j];
    float xn = (v[j] - mean) * inv * lw[c] + lb[c];
    lnx[base + c] = (_Float16)xn;
  }
}

// ---------------------------------------------------------------------------
// WMMA GEMM: C[M, ldc] (f32) = A[M, K] (f16, row-major) * W[N, K]^T (f16).
// One wave32 computes a (16*MT) x (16*NT) output strip. All operand loads for
// a k-step are issued as a batch before the MTxNT WMMA block so the compiler
// can use partial s_wait_loadcnt and overlap VMEM with the matrix pipe.
// Layouts per CDNA5 ISA 7.12.2 (16-bit A 16x32; B 32x16; 32-bit C/D 16x16).
// GUARD: predicate *stores* on n < Nw (loads never guarded; weight rows
// beyond Nw are zero-padded). RESID: fused residual add in epilogue.
// ---------------------------------------------------------------------------
template <int MT, int NT, bool GUARD, bool RESID>
__global__ void gemm_wmma_k(const _Float16* __restrict__ A,
                            const _Float16* __restrict__ W,
                            float* __restrict__ C,
                            const float* __restrict__ resid,
                            int K, int Nw, int ldc, int waves_per_row) {
  int wid  = blockIdx.x * (blockDim.x >> 5) + (threadIdx.x >> 5);
  int lane = threadIdx.x & 31;
  int m0 = (wid / waves_per_row) * (16 * MT);
  int n0 = (wid % waves_per_row) * (16 * NT);
  if (m0 >= M_TOK) return;

  int half = lane >> 4;   // 0: lanes 0-15, 1: lanes 16-31
  int l15  = lane & 15;

  v8f acc[MT][NT];
#pragma unroll
  for (int mt = 0; mt < MT; ++mt)
#pragma unroll
    for (int nt = 0; nt < NT; ++nt)
#pragma unroll
      for (int e = 0; e < 8; ++e) acc[mt][nt][e] = 0.f;

  // A row for this lane: row M = m0 + mt*16 + l15, K chunks at +8*half, +8*half+16
  const _Float16* arow0 = A + (size_t)(m0 + l15) * K + half * 8;
  // B row for this lane (per tile): weight row n, K chunks at +8*half, +8*half+16
  const _Float16* brow0 = W + (size_t)(n0 + l15) * K + half * 8;

  for (int k0 = 0; k0 < K; k0 += 32) {
    v16h av[MT];
    v16h bv[NT];
    // ---- batched operand loads (one clause; partial waits downstream) ----
#pragma unroll
    for (int mt = 0; mt < MT; ++mt) {
      const _Float16* ar = arow0 + (size_t)(mt * 16) * K + k0;
      v8h lo = *(const v8h*)(ar);
      v8h hi = *(const v8h*)(ar + 16);
#pragma unroll
      for (int e = 0; e < 8; ++e) { av[mt][e] = lo[e]; av[mt][e + 8] = hi[e]; }
    }
#pragma unroll
    for (int nt = 0; nt < NT; ++nt) {
      const _Float16* br = brow0 + (size_t)(nt * 16) * K + k0;
      v8h lo = *(const v8h*)(br);
      v8h hi = *(const v8h*)(br + 16);
#pragma unroll
      for (int e = 0; e < 8; ++e) { bv[nt][e] = lo[e]; bv[nt][e + 8] = hi[e]; }
    }
    // ---- matrix ops ----
#pragma unroll
    for (int mt = 0; mt < MT; ++mt)
#pragma unroll
      for (int nt = 0; nt < NT; ++nt)
        acc[mt][nt] = __builtin_amdgcn_wmma_f32_16x16x32_f16(
            /*neg_a=*/false, av[mt], /*neg_b=*/false, bv[nt],
            /*c_mod=*/(short)0, acc[mt][nt], /*reuse_a=*/false, /*reuse_b=*/false);
  }

  // Store: lane holds column n = n0+nt*16+l15, rows m0+mt*16 + 8*half + r
#pragma unroll
  for (int mt = 0; mt < MT; ++mt) {
#pragma unroll
    for (int nt = 0; nt < NT; ++nt) {
      int ng = n0 + nt * 16 + l15;
      if (GUARD && ng >= Nw) continue;
#pragma unroll
      for (int r = 0; r < 8; ++r) {
        int m = m0 + mt * 16 + half * 8 + r;
        size_t idx = (size_t)m * ldc + ng;
        float val = acc[mt][nt][r];
        if (RESID) val += resid[idx];
        C[idx] = val;
      }
    }
  }
}

// ---------------------------------------------------------------------------
// Depthwise causal conv1d (width 4) over L per batch + bias + SiLU.
// xs = xz[:, :256]; write f32 (scan) and f16 (x_proj WMMA A op).
// ---------------------------------------------------------------------------
__global__ void conv_silu_k(const float* __restrict__ xz,
                            const float* __restrict__ cw,  // [256,1,4]
                            const float* __restrict__ cb,
                            float* __restrict__ xsc, _Float16* __restrict__ xsch) {
  int i = blockIdx.x * 256 + threadIdx.x;   // over M_TOK * 256
  if (i >= M_TOK * DIN) return;
  int d   = i & (DIN - 1);
  int tok = i >> 8;
  int t = tok % Lseq;
  float acc = cb[d];
#pragma unroll
  for (int j = 0; j < 4; ++j) {
    int tt = t - 3 + j;
    if (tt >= 0) acc += cw[d * 4 + j] * xz[(size_t)(tok - 3 + j) * 512 + d];
  }
  float s = siluf(acc);
  xsc[i]  = s;
  xsch[i] = (_Float16)s;
}

// ---------------------------------------------------------------------------
// delta = softplus(dt @ dt_W^T + dt_b), K=8 -> plain VALU (too small for WMMA).
// ---------------------------------------------------------------------------
__global__ void delta_k(const float* __restrict__ dbl,
                        const float* __restrict__ dtW,  // [256,8]
                        const float* __restrict__ dtb,
                        float* __restrict__ delta) {
  int m = blockIdx.x;
  int d = threadIdx.x;
  const float* r = dbl + (size_t)m * NPROJ;
  float acc = dtb[d];
#pragma unroll
  for (int j = 0; j < DTR; ++j) acc += r[j] * dtW[d * DTR + j];
  float sp = (acc > 20.f) ? acc : log1pf(__expf(acc));
  delta[(size_t)m * DIN + d] = sp;
}

// ---------------------------------------------------------------------------
// Chunked selective scan.
// Wave layout: lane = 16*d_sub + n; wave handles 2 adjacent d channels x 16 n.
// Pass 1: chunk-local scan (h0=0) -> hloc[b,d,chunk,n], sumDelta[b,d,chunk].
// ---------------------------------------------------------------------------
__global__ void scan1_k(const float* __restrict__ delta, const float* __restrict__ dbl,
                        const float* __restrict__ xsc, const float* __restrict__ A_log,
                        float* __restrict__ hloc, float* __restrict__ sdelta) {
  int w    = blockIdx.x * (blockDim.x >> 5) + (threadIdx.x >> 5);
  int lane = threadIdx.x & 31;
  if (w >= Bb * 128 * NCHUNK) return;
  int b   = w / (128 * NCHUNK);
  int rem = w - b * (128 * NCHUNK);
  int dp  = rem / NCHUNK;
  int ch  = rem - dp * NCHUNK;
  int d = dp * 2 + (lane >> 4);
  int n = lane & 15;
  float Adn = -__expf(A_log[d * DST + n]);

  float h = 0.f, sd = 0.f;
  size_t tok0 = (size_t)b * Lseq + ch * CLEN;
  for (int s = 0; s < CLEN; ++s) {
    size_t tok = tok0 + s;
    float dt = delta[tok * DIN + d];
    float Bn = dbl[tok * NPROJ + DTR + n];
    float xv = xsc[tok * DIN + d];
    float dA = __expf(dt * Adn);
    h = dA * h + dt * Bn * xv;
    sd += dt;
  }
  size_t cidx = ((size_t)(b * DIN + d) * NCHUNK + ch);
  hloc[cidx * DST + n] = h;
  if (n == 0) sdelta[cidx] = sd;
}

// Pass 2: serial combine over chunks; one lane per (b,d,n). Writes carry-in
// state hin[b,d,chunk,n] for every chunk. (Pi exp(dt*A) = exp(A*Sum dt).)
__global__ void scan2_k(const float* __restrict__ A_log, const float* __restrict__ hloc,
                        const float* __restrict__ sdelta, float* __restrict__ hin) {
  int i = blockIdx.x * blockDim.x + threadIdx.x;
  if (i >= Bb * DIN * DST) return;
  int n = i & 15;
  int d = (i >> 4) & (DIN - 1);
  int b = i >> 12;
  float Adn = -__expf(A_log[d * DST + n]);
  size_t cb = (size_t)(b * DIN + d) * NCHUNK;
  float h = 0.f;
  for (int c = 0; c < NCHUNK; ++c) {
    hin[(cb + c) * DST + n] = h;
    float P = __expf(Adn * sdelta[cb + c]);
    h = P * h + hloc[(cb + c) * DST + n];
  }
}

// Pass 3: replay with carry-in, emit y_t = <h, C_t>, fuse +xs*D and *silu(z),
// store f16 A-operand for out-proj.
__global__ void scan3_k(const float* __restrict__ delta, const float* __restrict__ dbl,
                        const float* __restrict__ xsc, const float* __restrict__ xz,
                        const float* __restrict__ A_log, const float* __restrict__ Dp,
                        const float* __restrict__ hin, _Float16* __restrict__ yh) {
  int w    = blockIdx.x * (blockDim.x >> 5) + (threadIdx.x >> 5);
  int lane = threadIdx.x & 31;
  if (w >= Bb * 128 * NCHUNK) return;
  int b   = w / (128 * NCHUNK);
  int rem = w - b * (128 * NCHUNK);
  int dp  = rem / NCHUNK;
  int ch  = rem - dp * NCHUNK;
  int d = dp * 2 + (lane >> 4);
  int n = lane & 15;
  float Adn = -__expf(A_log[d * DST + n]);
  float Dd  = Dp[d];

  float h = hin[((size_t)(b * DIN + d) * NCHUNK + ch) * DST + n];
  size_t tok0 = (size_t)b * Lseq + ch * CLEN;
  for (int s = 0; s < CLEN; ++s) {
    size_t tok = tok0 + s;
    float dt = delta[tok * DIN + d];
    float Bn = dbl[tok * NPROJ + DTR + n];
    float Cn = dbl[tok * NPROJ + DTR + DST + n];
    float xv = xsc[tok * DIN + d];
    float dA = __expf(dt * Adn);
    h = dA * h + dt * Bn * xv;
    float y = h * Cn;
    // reduce over n (16-lane halves stay closed under xor masks 1,2,4,8)
    y += __shfl_xor(y, 8, 32);
    y += __shfl_xor(y, 4, 32);
    y += __shfl_xor(y, 2, 32);
    y += __shfl_xor(y, 1, 32);
    if (n == 0) {
      float zv = xz[tok * 512 + DIN + d];
      float out = (y + xv * Dd) * siluf(zv);
      yh[tok * DIN + d] = (_Float16)out;
    }
  }
}

// ---------------------------------------------------------------------------
// Final fold: out[b,c,s] = fused[b,s,c] + fused[b,s+HW,c]   (fused = proj+cat)
// ---------------------------------------------------------------------------
__global__ void fold_k(const float* __restrict__ fused, float* __restrict__ out) {
  int i = blockIdx.x * 256 + threadIdx.x;   // over Bb*HW*DIMC
  if (i >= Bb * HW * DIMC) return;
  int c = i % DIMC;
  int s = (i / DIMC) % HW;
  int b = i / (DIMC * HW);
  size_t r0 = ((size_t)b * Lseq + s) * DIMC + c;
  size_t r1 = ((size_t)b * Lseq + HW + s) * DIMC + c;
  out[((size_t)(b * DIMC) + c) * HW + s] = fused[r0] + fused[r1];
}

// ---------------------------------------------------------------------------
extern "C" void kernel_launch(void* const* d_in, const int* in_sizes, int n_in,
                              void* d_out, int out_size, void* d_ws, size_t ws_size,
                              hipStream_t stream) {
  const float* pan   = (const float*)d_in[0];
  const float* ms    = (const float*)d_in[1];
  const float* ln_w  = (const float*)d_in[2];
  const float* ln_b  = (const float*)d_in[3];
  const float* inW   = (const float*)d_in[4];   // [512,128]
  const float* convw = (const float*)d_in[5];   // [256,1,4]
  const float* convb = (const float*)d_in[6];
  const float* xpW   = (const float*)d_in[7];   // [40,256]
  const float* dtW   = (const float*)d_in[8];   // [256,8]
  const float* dtb   = (const float*)d_in[9];
  const float* Alog  = (const float*)d_in[10];  // [256,16]
  const float* Dpar  = (const float*)d_in[11];
  const float* outW  = (const float*)d_in[12];  // [128,256]
  float* outp = (float*)d_out;

  char* base = (char*)d_ws;
  size_t off = 0;
  auto alloc = [&](size_t bytes) -> char* {
    char* p = base + off;
    off = (off + bytes + 255) & ~(size_t)255;
    return p;
  };

  float*     cat_f32 = (float*)    alloc((size_t)M_TOK * DIMC * 4);
  _Float16*  lnx_f16 = (_Float16*) alloc((size_t)M_TOK * DIMC * 2);
  _Float16*  w_in    = (_Float16*) alloc((size_t)512 * 128 * 2);
  _Float16*  w_xp    = (_Float16*) alloc((size_t)NPAD * 256 * 2);  // zero-padded
  _Float16*  w_out   = (_Float16*) alloc((size_t)128 * 256 * 2);
  float*     xz      = (float*)    alloc((size_t)M_TOK * 512 * 4);
  float*     xsc     = (float*)    alloc((size_t)M_TOK * DIN * 4);
  _Float16*  xsch    = (_Float16*) alloc((size_t)M_TOK * DIN * 2);
  float*     dbl     = (float*)    alloc((size_t)M_TOK * NPROJ * 4);
  float*     delta   = (float*)    alloc((size_t)M_TOK * DIN * 4);
  _Float16*  y_f16   = (_Float16*) alloc((size_t)M_TOK * DIN * 2);
  float*     sdelta  = (float*)    alloc((size_t)Bb * DIN * NCHUNK * 4);
  float*     hloc    = (float*)    alloc((size_t)Bb * DIN * NCHUNK * DST * 4);
  float*     hin     = (float*)    alloc((size_t)Bb * DIN * NCHUNK * DST * 4);
  float*     fused   = (float*)    alloc((size_t)M_TOK * DIMC * 4);
  (void)ws_size; (void)in_sizes; (void)n_in; (void)out_size;

  // 1) weights -> f16 (x_proj padded to 48 zero rows)
  {
    int total = 512 * 128 + NPAD * 256 + 128 * 256;
    cvt_weights_k<<<(total + 255) / 256, 256, 0, stream>>>(inW, xpW, outW, w_in, w_xp, w_out);
  }
  // 2) cat + layernorm
  ln_cat_k<<<M_TOK / 8, 256, 0, stream>>>(pan, ms, ln_w, ln_b, cat_f32, lnx_f16);
  // 3) in_proj GEMM: xz[M,512] = lnx[M,128] * w_in[512,128]^T  (MT=2, NT=4)
  {
    int waves_per_row = 512 / 64;                       // 8
    int waves = (M_TOK / 32) * waves_per_row;           // 4608
    gemm_wmma_k<2, 4, false, false><<<waves / 8, 256, 0, stream>>>(
        lnx_f16, w_in, xz, nullptr, 128, 512, 512, waves_per_row);
  }
  // 4) conv + silu
  conv_silu_k<<<(M_TOK * DIN) / 256, 256, 0, stream>>>(xz, convw, convb, xsc, xsch);
  // 5) x_proj GEMM: dbl[M,40] = xsc[M,256] * w_xp[48,256]^T (store-guarded)
  {
    int waves = M_TOK / 32;                             // 576
    gemm_wmma_k<2, 3, true, false><<<waves / 8, 256, 0, stream>>>(
        xsch, w_xp, dbl, nullptr, 256, NPROJ, NPROJ, 1);
  }
  // 6) delta = softplus(dt @ dtW^T + dtb)
  delta_k<<<M_TOK, DIN, 0, stream>>>(dbl, dtW, dtb, delta);
  // 7) chunked selective scan
  {
    int waves = Bb * 128 * NCHUNK;                      // 24576
    scan1_k<<<waves / 8, 256, 0, stream>>>(delta, dbl, xsc, Alog, hloc, sdelta);
    scan2_k<<<(Bb * DIN * DST) / 256, 256, 0, stream>>>(Alog, hloc, sdelta, hin);
    scan3_k<<<waves / 8, 256, 0, stream>>>(delta, dbl, xsc, xz, Alog, Dpar, hin, y_f16);
  }
  // 8) out_proj GEMM + residual: fused[M,128] = y[M,256] * w_out[128,256]^T + cat
  {
    int waves_per_row = 128 / 64;                       // 2
    int waves = (M_TOK / 32) * waves_per_row;           // 1152
    gemm_wmma_k<2, 4, false, true><<<waves / 8, 256, 0, stream>>>(
        y_f16, w_out, fused, cat_f32, 256, 128, 128, waves_per_row);
  }
  // 9) fold p+m and transpose to [b,c,h,w]
  fold_k<<<(Bb * HW * DIMC) / 256, 256, 0, stream>>>(fused, outp);
}
